// GraphNativeEncoder_3728031613681
// MI455X (gfx1250) — compile-verified, matrix operand order checked by reading, not run on hardware
//
#include <hip/hip_runtime.h>
#include <hip/hip_bf16.h>

#define TT 64
#define HH 64
#define KNN 10

typedef __attribute__((ext_vector_type(16))) _Float16 v16h;
typedef __attribute__((ext_vector_type(8)))  float    v8f;

// ---------------------------------------------------------------------------
// WMMA helpers (CDNA5 16x16x32 f16 -> f32).
// A operand: lane l (0..15) = row M=l, K pairs {hi*8+2g, +1} in regs 0..3 and
// {16+hi*8+2g} in regs 4..7 where hi = l>>4 (ISA 7.12.2, 16-bit A 16x32).
// B operand for X @ W^T: lane holds column N of B = row N of W with the SAME
// per-lane K packing, so one loader serves both.
// ---------------------------------------------------------------------------
__device__ inline v16h wmma_load16(const _Float16* base, int ld, int lane) {
  int m  = lane & 15;
  int hi = (lane >> 4) << 3;          // 0 or 8
  const _Float16* row = base + m * ld;
  v16h r;
#pragma unroll
  for (int g = 0; g < 4; ++g) {
    r[2 * g]         = row[hi + 2 * g];
    r[2 * g + 1]     = row[hi + 2 * g + 1];
    r[8 + 2 * g]     = row[16 + hi + 2 * g];
    r[8 + 2 * g + 1] = row[16 + hi + 2 * g + 1];
  }
  return r;
}

__device__ inline v8f wmma_f32_16x16x32(v16h a, v16h b, v8f c) {
  return __builtin_amdgcn_wmma_f32_16x16x32_f16(false, a, false, b, (short)0, c,
                                                false, false);
}

// C/D layout: reg r, lanes 0-15 -> M=r, N=lane; lanes 16-31 -> M=8+r, N=lane-16.
__device__ inline void wmma_store(float* out, int ld, int lane, v8f acc, float bias) {
  int n  = lane & 15;
  int mb = (lane >> 4) * 8;
#pragma unroll
  for (int r = 0; r < 8; ++r) out[(mb + r) * ld + n] = acc[r] + bias;
}

// ---------------------------------------------------------------------------
// 1) x_agg = mean_t x; e = normalize(x_agg @ Wp^T)   (H -> 32)
// ---------------------------------------------------------------------------
__global__ __launch_bounds__(64) void k_node_embed(const float* __restrict__ x,
                                                   const float* __restrict__ Wp,
                                                   float* __restrict__ e) {
  int n = blockIdx.x;
  __shared__ float xa[HH];
  __shared__ float ev[32];
  __shared__ float inv;
  int h = threadIdx.x;
  const float* xb = x + (size_t)n * TT * HH;
  float s = 0.f;
  for (int t = 0; t < TT; ++t) s += xb[t * HH + h];
  xa[h] = s * (1.0f / TT);
  __syncthreads();
  if (h < 32) {
    const float* wr = Wp + h * HH;
    float acc = 0.f;
    for (int c = 0; c < HH; ++c) acc += xa[c] * wr[c];
    ev[h] = acc;
  }
  __syncthreads();
  if (h == 0) {
    float nn = 0.f;
    for (int c = 0; c < 32; ++c) nn += ev[c] * ev[c];
    inv = 1.0f / fmaxf(sqrtf(nn), 1e-12f);
  }
  __syncthreads();
  if (h < 32) e[n * 32 + h] = ev[h] * inv;
}

// ---------------------------------------------------------------------------
// 2) top-10 cosine neighbors per node (excl. self). One wave per node.
//    Ordering matches jax.lax.top_k: value desc, ties by ascending index.
// ---------------------------------------------------------------------------
__global__ __launch_bounds__(32) void k_topk(const float* __restrict__ e,
                                             float* __restrict__ vals,
                                             int* __restrict__ idx, int N) {
  int i = blockIdx.x;
  int lane = threadIdx.x;
  __shared__ float ei[32];
  __shared__ float cv[32 * KNN];
  __shared__ int   ci[32 * KNN];
  ei[lane] = e[i * 32 + lane];
  __syncthreads();
  float bv[KNN];
  int   bi[KNN];
#pragma unroll
  for (int k = 0; k < KNN; ++k) { bv[k] = -1e30f; bi[k] = 0x7fffffff; }
  for (int j = lane; j < N; j += 32) {
    if (j == i) continue;
    const float* ej = e + j * 32;
    float s = 0.f;
    for (int c = 0; c < 32; ++c) s += ei[c] * ej[c];
    if (s > bv[KNN - 1]) {             // strict: equal values keep earlier index
      int p = KNN - 1;
      while (p > 0 && bv[p - 1] < s) { bv[p] = bv[p - 1]; bi[p] = bi[p - 1]; --p; }
      bv[p] = s; bi[p] = j;
    }
  }
  for (int k = 0; k < KNN; ++k) { cv[lane * KNN + k] = bv[k]; ci[lane * KNN + k] = bi[k]; }
  __syncthreads();
  if (lane == 0) {
    for (int k = 0; k < KNN; ++k) {
      float best = -1e30f; int besti = 0x7fffffff; int bp = 0;
      for (int p = 0; p < 32 * KNN; ++p) {
        float v = cv[p]; int id = ci[p];
        if (v > best || (v == best && id < besti)) { best = v; besti = id; bp = p; }
      }
      cv[bp] = -1e31f;
      vals[i * KNN + k] = best;
      idx[i * KNN + k]  = besti;
    }
  }
}

// ---------------------------------------------------------------------------
// 3) temporal conv (kernel 3, pad 1) as 3 shifted 64x64 WMMA GEMMs per node:
//    xt[t,o] = b[o] + sum_k sum_i x[t+k-1,i] * w[o,i,k]
// ---------------------------------------------------------------------------
__global__ __launch_bounds__(256) void k_tconv(const float* __restrict__ x,
                                               const float* __restrict__ w,
                                               const float* __restrict__ b,
                                               float* __restrict__ xt) {
  int n = blockIdx.x;
  __shared__ _Float16 xp[(TT + 2) * HH];     // zero-padded rows 0 and 65
  __shared__ _Float16 wk[3][HH * HH];        // wk[k][o*64+i] = w[o,i,k]
  int tid = threadIdx.x;
  if (tid < HH) { xp[tid] = (_Float16)0.f; xp[(TT + 1) * HH + tid] = (_Float16)0.f; }
  const float* xb = x + (size_t)n * TT * HH;
  for (int i = tid; i < TT * HH; i += 256) xp[HH + i] = (_Float16)xb[i];
  for (int i = tid; i < HH * HH * 3; i += 256) {
    int o = i / (HH * 3), rem = i % (HH * 3), c = rem / 3, k = rem % 3;
    wk[k][o * HH + c] = (_Float16)w[i];
  }
  __syncthreads();
  int wave = tid >> 5, lane = tid & 31;
#pragma unroll
  for (int it = 0; it < 2; ++it) {
    int tile = wave * 2 + it;
    int tm = tile >> 2, tn = tile & 3;
    v8f acc = {};
#pragma unroll
    for (int k = 0; k < 3; ++k)
#pragma unroll
      for (int kk = 0; kk < HH; kk += 32) {
        v16h a  = wmma_load16(&xp[(tm * 16 + k) * HH + kk], HH, lane);
        v16h bb = wmma_load16(&wk[k][(tn * 16) * HH + kk], HH, lane);
        acc = wmma_f32_16x16x32(a, bb, acc);
      }
    float bias = b[tn * 16 + (lane & 15)];
    wmma_store(xt + (size_t)n * TT * HH + (tm * 16) * HH + tn * 16, HH, lane, acc, bias);
  }
}

// ---------------------------------------------------------------------------
// 4) per-node GEMM: out[n] = act[n](64x64) @ W^T + bias   (WMMA)
// ---------------------------------------------------------------------------
__global__ __launch_bounds__(256) void k_gemm_nodes(const float* __restrict__ act,
                                                    const float* __restrict__ W,
                                                    const float* __restrict__ bias,
                                                    float* __restrict__ out) {
  int n = blockIdx.x;
  __shared__ _Float16 As[TT * HH];
  __shared__ _Float16 Ws[HH * HH];
  int tid = threadIdx.x;
  const float* ab = act + (size_t)n * TT * HH;
  for (int i = tid; i < TT * HH; i += 256) { As[i] = (_Float16)ab[i]; Ws[i] = (_Float16)W[i]; }
  __syncthreads();
  int wave = tid >> 5, lane = tid & 31;
#pragma unroll
  for (int it = 0; it < 2; ++it) {
    int tile = wave * 2 + it;
    int tm = tile >> 2, tn = tile & 3;
    v8f acc = {};
#pragma unroll
    for (int kk = 0; kk < HH; kk += 32) {
      v16h a  = wmma_load16(&As[(tm * 16) * HH + kk], HH, lane);
      v16h bb = wmma_load16(&Ws[(tn * 16) * HH + kk], HH, lane);
      acc = wmma_f32_16x16x32(a, bb, acc);
    }
    float bv = bias[tn * 16 + (lane & 15)];
    wmma_store(out + (size_t)n * TT * HH + (tm * 16) * HH + tn * 16, HH, lane, acc, bv);
  }
}

// ---------------------------------------------------------------------------
// 5) attention dot products per (n,t): asrc/adst (biases folded in)
// ---------------------------------------------------------------------------
__global__ __launch_bounds__(256) void k_attdots(const float* __restrict__ xt,
                                                 const float* __restrict__ ws,
                                                 const float* __restrict__ wd,
                                                 const float* __restrict__ bs,
                                                 const float* __restrict__ bd,
                                                 float* __restrict__ asrc,
                                                 float* __restrict__ adst, int NT) {
  int i = blockIdx.x * blockDim.x + threadIdx.x;
  if (i >= NT) return;
  const float* row = xt + (size_t)i * HH;
  float s1 = 0.f, s2 = 0.f;
  for (int h = 0; h < HH; ++h) { float v = row[h]; s1 += v * ws[h]; s2 += v * wd[h]; }
  asrc[i] = s1 + bs[0];
  adst[i] = s2 + bd[0];
}

// ---------------------------------------------------------------------------
// 6) edge build (fixed + kNN, sigmoid mix) and per-dst histogram
// ---------------------------------------------------------------------------
__global__ __launch_bounds__(256) void k_zero(int* __restrict__ p, int n) {
  int i = blockIdx.x * blockDim.x + threadIdx.x;
  if (i < n) p[i] = 0;
}

__global__ __launch_bounds__(256) void k_edges(const int* __restrict__ fei,
                                               const float* __restrict__ fea,
                                               const float* __restrict__ mix,
                                               const int* __restrict__ knn_idx,
                                               const float* __restrict__ knn_val,
                                               int* __restrict__ src, int* __restrict__ dst,
                                               float* __restrict__ ea, int* __restrict__ counts,
                                               int Ef, int Et) {
  int e = blockIdx.x * blockDim.x + threadIdx.x;
  if (e >= Et) return;
  float alpha = 1.f / (1.f + expf(-mix[0]));
  int s, d; float w;
  if (e < Ef) {
    s = fei[e]; d = fei[Ef + e]; w = fea[e] * (1.f - alpha);
  } else {
    int r = e - Ef, i = r / KNN, k = r % KNN;
    s = i; d = knn_idx[i * KNN + k]; w = knn_val[i * KNN + k] * alpha;
  }
  src[e] = s; dst[e] = d; ea[e] = w;
  atomicAdd(&counts[d], 1);
}

__global__ void k_scan(const int* __restrict__ counts, int* __restrict__ offsets, int N) {
  if (threadIdx.x == 0 && blockIdx.x == 0) {
    int acc = 0;
    for (int i = 0; i < N; ++i) { offsets[i] = acc; acc += counts[i]; }
    offsets[N] = acc;
  }
}

// deterministic CSR fill: edges kept in ascending edge-id order per segment
// (matches jax segment_* accumulation order; no atomics -> replay-stable)
__global__ __launch_bounds__(256) void k_fill(const int* __restrict__ dst,
                                              const int* __restrict__ offsets,
                                              int* __restrict__ elist, int E, int N) {
  int d = blockIdx.x * blockDim.x + threadIdx.x;
  if (d >= N) return;
  int pos = offsets[d];
  for (int e = 0; e < E; ++e)
    if (dst[e] == d) elist[pos++] = e;
}

// ---------------------------------------------------------------------------
// 7) per-destination softmax + weighted gather:  out[d] += sum_e coef_e * y[src_e]
// ---------------------------------------------------------------------------
__global__ __launch_bounds__(256) void k_aggr(const float* __restrict__ y,
                                              const float* __restrict__ asrc,
                                              const float* __restrict__ adst,
                                              const int* __restrict__ src,
                                              const float* __restrict__ ea,
                                              const int* __restrict__ offsets,
                                              const int* __restrict__ elist,
                                              float* __restrict__ out) {
  int d = blockIdx.x;
  __shared__ float ad[TT], mm[TT], zz[TT];
  int tid = threadIdx.x;
  int beg = offsets[d], end = offsets[d + 1];
  if (tid < TT) ad[tid] = adst[d * TT + tid];
  __syncthreads();
  if (tid < TT) {
    int t = tid;
    float m = -1e30f;
    for (int j = beg; j < end; ++j) {
      int e = elist[j], s = src[e];
      float a = asrc[s * TT + t] + ad[t];
      a = a >= 0.f ? a : 0.2f * a;
      m = fmaxf(m, a);
    }
    mm[t] = m;
    float z = 0.f;
    for (int j = beg; j < end; ++j) {
      int e = elist[j], s = src[e];
      float a = asrc[s * TT + t] + ad[t];
      a = a >= 0.f ? a : 0.2f * a;
      z += expf(a - m);
    }
    zz[t] = z;
  }
  __syncthreads();
  int t  = tid >> 2;
  int h0 = (tid & 3) * 16;
  float acc[16];
#pragma unroll
  for (int q = 0; q < 16; ++q) acc[q] = 0.f;
  float m = mm[t], zinv = 1.f / (zz[t] + 1e-16f), adt = ad[t];
  for (int j = beg; j < end; ++j) {
    int e = elist[j], s = src[e];
    float a = asrc[s * TT + t] + adt;
    a = a >= 0.f ? a : 0.2f * a;
    float c = ea[e] * expf(a - m) * zinv;
    const float4* yr = (const float4*)(y + ((size_t)s * TT + t) * HH + h0);
#pragma unroll
    for (int q = 0; q < 4; ++q) {
      float4 v = yr[q];
      acc[4 * q + 0] += c * v.x; acc[4 * q + 1] += c * v.y;
      acc[4 * q + 2] += c * v.z; acc[4 * q + 3] += c * v.w;
    }
  }
  float* op = out + ((size_t)d * TT + t) * HH + h0;
#pragma unroll
  for (int q = 0; q < 16; ++q) op[q] += acc[q];
}

// ---------------------------------------------------------------------------
extern "C" void kernel_launch(void* const* d_in, const int* in_sizes, int n_in,
                              void* d_out, int out_size, void* d_ws, size_t ws_size,
                              hipStream_t stream) {
  const float* x        = (const float*)d_in[0];
  const int*   fei      = (const int*)d_in[1];
  const float* fea      = (const float*)d_in[2];
  const float* Wp       = (const float*)d_in[3];
  const float* mix      = (const float*)d_in[4];
  const float* tconv_w  = (const float*)d_in[5];
  const float* tconv_b  = (const float*)d_in[6];
  const float* Wmsg     = (const float*)d_in[7];
  const float* bmsg     = (const float*)d_in[8];
  const float* Wself    = (const float*)d_in[9];
  const float* bself    = (const float*)d_in[10];
  const float* att_sw   = (const float*)d_in[11];
  const float* att_sb   = (const float*)d_in[12];
  const float* att_dw   = (const float*)d_in[13];
  const float* att_db   = (const float*)d_in[14];
  float* out = (float*)d_out;

  const int N  = in_sizes[0] / (TT * HH);
  const int Ef = in_sizes[1] / 2;
  const int Et = Ef + N * KNN;
  const int NT = N * TT;

  char* p = (char*)d_ws;
  auto alloc = [&](size_t bytes) -> void* {
    void* r = (void*)p;
    p += (bytes + 255) & ~(size_t)255;
    return r;
  };
  float* e_emb   = (float*)alloc((size_t)N * 32 * 4);
  float* knn_val = (float*)alloc((size_t)N * KNN * 4);
  int*   knn_idx = (int*)  alloc((size_t)N * KNN * 4);
  float* xt      = (float*)alloc((size_t)N * TT * HH * 4);
  float* y       = (float*)alloc((size_t)N * TT * HH * 4);
  float* asrc    = (float*)alloc((size_t)NT * 4);
  float* adst    = (float*)alloc((size_t)NT * 4);
  int*   srcA    = (int*)  alloc((size_t)Et * 4);
  int*   dstA    = (int*)  alloc((size_t)Et * 4);
  float* eaA     = (float*)alloc((size_t)Et * 4);
  int*   counts  = (int*)  alloc((size_t)N * 4);
  int*   offsets = (int*)  alloc((size_t)(N + 1) * 4);
  int*   elist   = (int*)  alloc((size_t)Et * 4);

  k_node_embed<<<N, 64, 0, stream>>>(x, Wp, e_emb);
  k_topk<<<N, 32, 0, stream>>>(e_emb, knn_val, knn_idx, N);
  k_tconv<<<N, 256, 0, stream>>>(x, tconv_w, tconv_b, xt);
  k_gemm_nodes<<<N, 256, 0, stream>>>(xt, Wmsg, bmsg, y);     // y = msg projection
  k_gemm_nodes<<<N, 256, 0, stream>>>(x, Wself, bself, out);  // out = self projection
  k_attdots<<<(NT + 255) / 256, 256, 0, stream>>>(xt, att_sw, att_dw, att_sb, att_db,
                                                  asrc, adst, NT);
  k_zero<<<(N + 255) / 256, 256, 0, stream>>>(counts, N);
  k_edges<<<(Et + 255) / 256, 256, 0, stream>>>(fei, fea, mix, knn_idx, knn_val,
                                                srcA, dstA, eaA, counts, Ef, Et);
  k_scan<<<1, 32, 0, stream>>>(counts, offsets, N);
  k_fill<<<(N + 255) / 256, 256, 0, stream>>>(dstA, offsets, elist, Et, N);
  k_aggr<<<N, 256, 0, stream>>>(y, asrc, adst, srcA, eaA, offsets, elist, out);
}
